// GPTDecoder_89515708383307
// MI455X (gfx1250) — matrix operand order repaired and checked by reading, not compiled, not run on hardware
//
#include <hip/hip_runtime.h>
#include <hip/hip_bf16.h>

// ---------------------------------------------------------------------------
// GPT decoder forward for MI455X (gfx1250, wave32, WMMA).
// GEMM: 64x64 block tile (4 waves x 16 rows), K-step 32, double-buffered LDS.
//   pipeline per K-step:  global_load_b128 (next tile, stays in flight)
//                         -> ds_load_b128 fragments + v_wmma_f32_16x16x32_f16
//                         -> cvt f32->f16 + ds_store (next tile, other buffer)
//   one s_barrier per K-step; LDS rows padded to 40 halves (conflict-free).
//   TB/GELU are template params -> branch-free inner loop.
// Workload is HBM-bound on attention-map + logits writes; the per-layer 64MB
// attention slice stays L2-resident (192MB) for the A*V re-read.
// ---------------------------------------------------------------------------

typedef __attribute__((ext_vector_type(16))) _Float16 v16h;
typedef __attribute__((ext_vector_type(8)))  float    v8f;

#define BM 64
#define BN 64
#define BK 32
#define LDSTRIDE 40   // halves per padded LDS row (80B -> 16 rows hit distinct banks)

struct TileRegs { float4 a[4]; float4 b[4]; };

// Issue global loads for one 64x32 A tile + 32x64 (or 64x32^T) B tile.
// Fixed 4 passes x 128 threads x float4: fully coalesced, no exec masking.
template <bool TB>
__device__ __forceinline__ void tile_load(TileRegs& t,
    const float* __restrict__ A, int lda,
    const float* __restrict__ Bm, int ldb,
    int mb0, int nb0, int k0, int tid)
{
#pragma unroll
    for (int p = 0; p < 4; ++p) {
        const int idx = tid * 4 + p * 512;
        {
            const int r = idx / BK, c = idx % BK;
            const float* src = A + (size_t)(mb0 + r) * lda + (k0 + c);
            t.a[p] = *(const float4*)src;
            __builtin_prefetch(src + BK, 0, 3);       // speculative: safe OOB
        }
        if (TB) {   // B[N,K] row-major: contiguous along K
            const int n = idx / BK, c = idx % BK;
            const float* src = Bm + (size_t)(nb0 + n) * ldb + (k0 + c);
            t.b[p] = *(const float4*)src;
            __builtin_prefetch(src + BK, 0, 3);
        } else {    // B[K,N] row-major: contiguous along N
            const int kk = idx / BN, n = idx % BN;
            const float* src = Bm + (size_t)(k0 + kk) * ldb + (nb0 + n);
            t.b[p] = *(const float4*)src;
            __builtin_prefetch(src + (size_t)BK * ldb, 0, 3);
        }
    }
}

// Convert f32->f16 and store the tile into LDS (As[m][k], Bs[n][k]).
template <bool TB>
__device__ __forceinline__ void tile_store(const TileRegs& t,
    _Float16* __restrict__ Asb, _Float16* __restrict__ Bsb, int tid)
{
#pragma unroll
    for (int p = 0; p < 4; ++p) {
        const int idx = tid * 4 + p * 512;
        {
            const int r = idx / BK, c = idx % BK;
            _Float16* d = &Asb[r * LDSTRIDE + c];
            d[0] = (_Float16)t.a[p].x; d[1] = (_Float16)t.a[p].y;
            d[2] = (_Float16)t.a[p].z; d[3] = (_Float16)t.a[p].w;
        }
        if (TB) {
            const int n = idx / BK, c = idx % BK;
            _Float16* d = &Bsb[n * LDSTRIDE + c];
            d[0] = (_Float16)t.b[p].x; d[1] = (_Float16)t.b[p].y;
            d[2] = (_Float16)t.b[p].z; d[3] = (_Float16)t.b[p].w;
        } else {    // scatter float4-along-N into Bs[n][k]
            const int kk = idx / BN, n = idx % BN;
            Bsb[(n + 0) * LDSTRIDE + kk] = (_Float16)t.b[p].x;
            Bsb[(n + 1) * LDSTRIDE + kk] = (_Float16)t.b[p].y;
            Bsb[(n + 2) * LDSTRIDE + kk] = (_Float16)t.b[p].z;
            Bsb[(n + 3) * LDSTRIDE + kk] = (_Float16)t.b[p].w;
        }
    }
}

// TB: B stored transposed ([N,K] row-major); GELU: exact-GELU epilogue
// C = alpha * A*B + bias  (bias may be null)
// batch z: offset = (z / bmod) * s?1 + (z % bmod) * s?2
template <bool TB, bool GELU>
__global__ void __launch_bounds__(128) wmma_gemm_f16(
    const float* __restrict__ A, int lda, long long sA1, long long sA2,
    const float* __restrict__ Bm, int ldb, long long sB1, long long sB2,
    const float* __restrict__ bias,
    float* __restrict__ C, int ldc, long long sC1, long long sC2,
    int M, int N, int K, float alpha, int bmod)
{
    __shared__ _Float16 As[2][BM * LDSTRIDE];   // [m][k], k in 0..31
    __shared__ _Float16 Bs[2][BN * LDSTRIDE];   // [n][k] (B transposed into LDS)

    const int z = blockIdx.z;
    A  += (long long)(z / bmod) * sA1 + (long long)(z % bmod) * sA2;
    Bm += (long long)(z / bmod) * sB1 + (long long)(z % bmod) * sB2;
    C  += (long long)(z / bmod) * sC1 + (long long)(z % bmod) * sC2;

    const int tid  = threadIdx.x;
    const int lane = tid & 31;
    const int wave = tid >> 5;
    const int mb0  = blockIdx.y * BM;
    const int nb0  = blockIdx.x * BN;

    // --- lane mapping used here for 16-bit WMMA fragments (contiguous k; A and B match) ---
    // A (16x32): row m = lane&15; lanes 0-15 hold K 0..15, lanes
    //            16-31 hold K 16..31; 2 consecutive K per VGPR.
    const int am  = wave * 16 + (lane & 15);
    const int akb = (lane >> 4) * 16;
    // B (32x16): col n = lane&15; lanes 0-15 hold K 0..15, lanes 16-31 K 16..31
    const int bn  = lane & 15;
    const int bkb = (lane >> 4) * 16;

    v8f acc[4];
#pragma unroll
    for (int j = 0; j < 4; ++j)
#pragma unroll
        for (int i = 0; i < 8; ++i) acc[j][i] = 0.0f;

    // prologue: stage tile 0 into buffer 0
    TileRegs t;
    tile_load<TB>(t, A, lda, Bm, ldb, mb0, nb0, 0, tid);
    tile_store<TB>(t, As[0], Bs[0], tid);

    const int nsteps = K / BK;
    for (int i = 0; i < nsteps; ++i) {
        __syncthreads();                       // buffer (i&1) ready block-wide
        const int  cur  = i & 1;
        const bool more = (i + 1) < nsteps;

        // issue next tile's global loads; they stay in flight during the WMMAs
        if (more) tile_load<TB>(t, A, lda, Bm, ldb, mb0, nb0, (i + 1) * BK, tid);

        // ---- fragments from LDS (merged to ds_load_b128 by contiguity) ----
        v16h afrag;
        const _Float16* arow = &As[cur][am * LDSTRIDE];
#pragma unroll
        for (int v = 0; v < 8; ++v) {
            const int kk = akb + 2 * v;
            afrag[2 * v]     = arow[kk];
            afrag[2 * v + 1] = arow[kk + 1];
        }
#pragma unroll
        for (int j = 0; j < 4; ++j) {
            const _Float16* brow = &Bs[cur][(j * 16 + bn) * LDSTRIDE];
            v16h bfrag;
#pragma unroll
            for (int v = 0; v < 8; ++v) {
                bfrag[2 * v]     = brow[bkb + 2 * v];
                bfrag[2 * v + 1] = brow[bkb + 2 * v + 1];
            }
            acc[j] = __builtin_amdgcn_wmma_f32_16x16x32_f16(
                false, afrag, false, bfrag, (short)0, acc[j], false, false);
        }

        // drain the in-flight loads, convert, fill the other buffer
        if (more) tile_store<TB>(t, As[cur ^ 1], Bs[cur ^ 1], tid);
    }

    // epilogue: C/D layout -> lane n = lane&15, VGPR v holds m = v + 8*(lane>=16)
    const int cm = mb0 + wave * 16 + ((lane >> 4) * 8);
#pragma unroll
    for (int j = 0; j < 4; ++j) {
        const int n  = nb0 + j * 16 + (lane & 15);
        const float bv = bias ? bias[n] : 0.0f;
#pragma unroll
        for (int v = 0; v < 8; ++v) {
            float val = acc[j][v] * alpha + bv;
            if (GELU)   // exact GELU: 0.5*x*(1+erf(x/sqrt(2)))
                val = 0.5f * val * (1.0f + erff(val * 0.70710678118654752f));
            C[(size_t)(cm + v) * ldc + n] = val;
        }
    }
}

// x[bs,d] = tok_emb[tokens[bs], d] * sqrt(D) + PE(s,d)
__global__ void embed_kernel(const int* __restrict__ tokens,
                             const float* __restrict__ emb,
                             float* __restrict__ x, int S, int D)
{
    const int bs = blockIdx.x;
    const int s  = bs % S;
    const int t  = tokens[bs];
    const float sqrtD = 22.62741699796952f;           // sqrt(512)
    const float c = -9.210340371976184f / (float)D;   // -ln(10000)/D
    for (int d = threadIdx.x; d < D; d += blockDim.x) {
        const int i = d & ~1;                         // 2*(d/2)
        const float ang = (float)s * __expf((float)i * c);
        const float pe  = (d & 1) ? __cosf(ang) : __sinf(ang);
        x[(size_t)bs * D + d] = emb[(size_t)t * D + d] * sqrtD + pe;
    }
}

// causal softmax in place over rows of [B*H*S, S]; masked cols -> 0
__global__ void softmax_causal(float* __restrict__ attn, int S)
{
    const int row = blockIdx.x;
    const int s   = row % S;
    float* p = attn + (size_t)row * S;
    __shared__ float red[256];
    const int tid = threadIdx.x;

    float mx = -3.0e38f;
    for (int t = tid; t <= s; t += blockDim.x) mx = fmaxf(mx, p[t]);
    red[tid] = mx; __syncthreads();
    for (int o = 128; o > 0; o >>= 1) {
        if (tid < o) red[tid] = fmaxf(red[tid], red[tid + o]);
        __syncthreads();
    }
    mx = red[0]; __syncthreads();

    float sum = 0.0f;
    for (int t = tid; t <= s; t += blockDim.x) {
        const float e = __expf(p[t] - mx);
        p[t] = e; sum += e;
    }
    red[tid] = sum; __syncthreads();
    for (int o = 128; o > 0; o >>= 1) {
        if (tid < o) red[tid] += red[tid + o];
        __syncthreads();
    }
    const float inv = 1.0f / red[0];
    for (int t = tid; t < S; t += blockDim.x)
        p[t] = (t <= s) ? p[t] * inv : 0.0f;
}

// out = LayerNorm(x + res) * g + b   (res may be null; out may alias x)
__global__ void add_layernorm(const float* __restrict__ x,
                              const float* __restrict__ res,
                              const float* __restrict__ g,
                              const float* __restrict__ b,
                              float* __restrict__ out, int D)
{
    const int row = blockIdx.x;
    const float* xp = x + (size_t)row * D;
    const float* rp = res ? res + (size_t)row * D : nullptr;
    __shared__ float r1[256];
    __shared__ float r2[256];
    const int tid = threadIdx.x;

    float s = 0.0f, s2 = 0.0f;
    for (int d = tid; d < D; d += blockDim.x) {
        const float v = xp[d] + (rp ? rp[d] : 0.0f);
        s += v; s2 += v * v;
    }
    r1[tid] = s; r2[tid] = s2; __syncthreads();
    for (int o = 128; o > 0; o >>= 1) {
        if (tid < o) { r1[tid] += r1[tid + o]; r2[tid] += r2[tid + o]; }
        __syncthreads();
    }
    const float mean = r1[0] / (float)D;
    const float var  = r2[0] / (float)D - mean * mean;
    const float rstd = rsqrtf(var + 1e-5f);
    for (int d = tid; d < D; d += blockDim.x) {
        const float v = xp[d] + (rp ? rp[d] : 0.0f);
        out[(size_t)row * D + d] = (v - mean) * rstd * g[d] + b[d];
    }
}

// ---------------------------------------------------------------------------

static void launch_gemm(hipStream_t st,
                        const float* A, int lda, long long sA1, long long sA2,
                        const float* B, int ldb, long long sB1, long long sB2,
                        const float* bias,
                        float* C, int ldc, long long sC1, long long sC2,
                        int M, int N, int K, float alpha, int flags,
                        int bmod, int batches)
{
    dim3 grid(N / BN, M / BM, batches);
    const bool tB   = (flags & 1) != 0;
    const bool gelu = (flags & 2) != 0;
    if (tB) {
        if (gelu) wmma_gemm_f16<true, true><<<grid, 128, 0, st>>>(
            A, lda, sA1, sA2, B, ldb, sB1, sB2, bias, C, ldc, sC1, sC2, M, N, K, alpha, bmod);
        else      wmma_gemm_f16<true, false><<<grid, 128, 0, st>>>(
            A, lda, sA1, sA2, B, ldb, sB1, sB2, bias, C, ldc, sC1, sC2, M, N, K, alpha, bmod);
    } else {
        if (gelu) wmma_gemm_f16<false, true><<<grid, 128, 0, st>>>(
            A, lda, sA1, sA2, B, ldb, sB1, sB2, bias, C, ldc, sC1, sC2, M, N, K, alpha, bmod);
        else      wmma_gemm_f16<false, false><<<grid, 128, 0, st>>>(
            A, lda, sA1, sA2, B, ldb, sB1, sB2, bias, C, ldc, sC1, sC2, M, N, K, alpha, bmod);
    }
}

extern "C" void kernel_launch(void* const* d_in, const int* in_sizes, int n_in,
                              void* d_out, int out_size, void* d_ws, size_t ws_size,
                              hipStream_t stream)
{
    constexpr int Bn = 2, S = 1024, D = 512, H = 8, L = 6, DFF = 2048, V = 32000;
    constexpr int DK = D / H;
    constexpr int M  = Bn * S;                 // 2048 token rows

    const int*   tokens  = (const int*)  d_in[0];
    const float* tok_emb = (const float*)d_in[1];
    const float* Wq  = (const float*)d_in[2];
    const float* bq  = (const float*)d_in[3];
    const float* Wk  = (const float*)d_in[4];
    const float* bk  = (const float*)d_in[5];
    const float* Wv  = (const float*)d_in[6];
    const float* bv  = (const float*)d_in[7];
    const float* Wo  = (const float*)d_in[8];
    const float* bo  = (const float*)d_in[9];
    const float* ln1g = (const float*)d_in[10];
    const float* ln1b = (const float*)d_in[11];
    const float* W1  = (const float*)d_in[12];
    const float* b1  = (const float*)d_in[13];
    const float* W2  = (const float*)d_in[14];
    const float* b2  = (const float*)d_in[15];
    const float* ln2g = (const float*)d_in[16];
    const float* ln2b = (const float*)d_in[17];
    const float* lnfg = (const float*)d_in[18];
    const float* lnfb = (const float*)d_in[19];
    const float* Wout = (const float*)d_in[20];
    const float* bout = (const float*)d_in[21];

    // d_out = [logits (B,S,V) | attns (L,B,H,S,S)], f32
    float* logits = (float*)d_out;
    float* attns  = logits + (size_t)Bn * S * V;

    // workspace layout (floats)
    float* ws  = (float*)d_ws;
    float* x   = ws;                            // [M, D]
    float* q   = x   + (size_t)M * D;           // [M, D]
    float* k   = q   + (size_t)M * D;           // [M, D]
    float* v   = k   + (size_t)M * D;           // [M, D]
    float* ctx = v   + (size_t)M * D;           // [M, D]
    float* tmp = ctx + (size_t)M * D;           // [M, D]
    float* hbf = tmp + (size_t)M * D;           // [M, DFF]
    float* xln = hbf + (size_t)M * DFF;         // [M, D]

    // 1) embedding + positional encoding
    embed_kernel<<<M, 256, 0, stream>>>(tokens, tok_emb, x, S, D);

    const float inv_scale = 0.125f;             // 1/sqrt(DK)

    for (int l = 0; l < L; ++l) {
        const float* Wql = Wq + (size_t)l * D * D;
        const float* Wkl = Wk + (size_t)l * D * D;
        const float* Wvl = Wv + (size_t)l * D * D;
        const float* Wol = Wo + (size_t)l * D * D;
        const float* W1l = W1 + (size_t)l * D * DFF;
        const float* W2l = W2 + (size_t)l * DFF * D;
        float* attnL = attns + (size_t)l * Bn * H * S * S;

        // 2) QKV projections: [M,D] x [D,D]
        launch_gemm(stream, x, D, 0, 0, Wql, D, 0, 0, bq + (size_t)l * D,
                    q, D, 0, 0, M, D, D, 1.0f, 0, 1, 1);
        launch_gemm(stream, x, D, 0, 0, Wkl, D, 0, 0, bk + (size_t)l * D,
                    k, D, 0, 0, M, D, D, 1.0f, 0, 1, 1);
        launch_gemm(stream, x, D, 0, 0, Wvl, D, 0, 0, bv + (size_t)l * D,
                    v, D, 0, 0, M, D, D, 1.0f, 0, 1, 1);

        // 3) scores = (1/8) * q @ k^T per (b,h), written straight into d_out
        //    batch z = b*H + h: A/B offset = b*S*D + h*DK, C offset = z*S*S
        launch_gemm(stream,
                    q, D, (long long)S * D, DK,
                    k, D, (long long)S * D, DK,
                    nullptr,
                    attnL, S, (long long)H * S * S, (long long)S * S,
                    S, S, DK, inv_scale, /*transB*/1, H, Bn * H);

        // 4) causal softmax in place (masked entries -> 0)
        softmax_causal<<<Bn * H * S, 256, 0, stream>>>(attnL, S);

        // 5) ctx = attn @ v per (b,h)  (attn slice is L2-resident: 64MB < 192MB)
        launch_gemm(stream,
                    attnL, S, (long long)H * S * S, (long long)S * S,
                    v, D, (long long)S * D, DK,
                    nullptr,
                    ctx, D, (long long)S * D, DK,
                    S, DK, S, 1.0f, 0, H, Bn * H);

        // 6) attn_out = ctx @ Wo + bo
        launch_gemm(stream, ctx, D, 0, 0, Wol, D, 0, 0, bo + (size_t)l * D,
                    tmp, D, 0, 0, M, D, D, 1.0f, 0, 1, 1);

        // 7) x = LN(x + attn_out)
        add_layernorm<<<M, 256, 0, stream>>>(x, tmp,
                                             ln1g + (size_t)l * D,
                                             ln1b + (size_t)l * D, x, D);

        // 8) h = gelu(x @ W1 + b1)   (exact erf GELU fused in epilogue)
        launch_gemm(stream, x, D, 0, 0, W1l, DFF, 0, 0, b1 + (size_t)l * DFF,
                    hbf, DFF, 0, 0, M, DFF, D, 1.0f, /*gelu*/2, 1, 1);

        // 9) ff = h @ W2 + b2
        launch_gemm(stream, hbf, DFF, 0, 0, W2l, D, 0, 0, b2 + (size_t)l * D,
                    tmp, D, 0, 0, M, D, DFF, 1.0f, 0, 1, 1);

        // 10) x = LN(x + ff)
        add_layernorm<<<M, 256, 0, stream>>>(x, tmp,
                                             ln2g + (size_t)l * D,
                                             ln2b + (size_t)l * D, x, D);
    }

    // final LN then logits = xln @ Wout + bout  -> d_out
    add_layernorm<<<M, 256, 0, stream>>>(x, nullptr, lnfg, lnfb, xln, D);
    launch_gemm(stream, xln, D, 0, 0, Wout, V, 0, 0, bout,
                logits, V, 0, 0, M, V, D, 1.0f, 0, 1, 1);
}
